// BitNetV3MLP_10419590660566
// MI455X (gfx1250) — 8-bit matrix operand order repaired and checked by reading, compiled, not run on hardware
//
#include <hip/hip_runtime.h>
#include <hip/hip_bf16.h>
#include <stdint.h>

typedef __attribute__((ext_vector_type(8))) int v8i;
typedef __attribute__((ext_vector_type(4))) int v4i;

#define HDIM 2048
#define IDIM 8192
#define EPSQ 1e-5f
#define TPAD 80

__device__ __forceinline__ void async_copy_b128(const int8_t* g, int8_t* lds) {
#if __has_builtin(__builtin_amdgcn_global_load_async_to_lds_b128)
  typedef __attribute__((address_space(1))) v4i* gp_t;
  typedef __attribute__((address_space(3))) v4i* lp_t;
  gp_t gsrc = (gp_t)(int8_t*)g;
  lp_t ldst = (lp_t)lds;
  __builtin_amdgcn_global_load_async_to_lds_b128(gsrc, ldst, 0, 0);
#else
  const uint32_t lds_off =
      (uint32_t)(uintptr_t)(__attribute__((address_space(3))) int8_t*)lds;
  const uint64_t gaddr = (uint64_t)(uintptr_t)g;
  asm volatile("global_load_async_to_lds_b128 %0, %1, off"
               :
               : "v"(lds_off), "v"(gaddr)
               : "memory");
#endif
}

__device__ __forceinline__ void wait_async_le6() {
#if __has_builtin(__builtin_amdgcn_s_wait_asynccnt)
  __builtin_amdgcn_s_wait_asynccnt(6);
#else
  asm volatile("s_wait_asynccnt 0x6" ::: "memory");
#endif
}
__device__ __forceinline__ void wait_async_le0() {
#if __has_builtin(__builtin_amdgcn_s_wait_asynccnt)
  __builtin_amdgcn_s_wait_asynccnt(0);
#else
  asm volatile("s_wait_asynccnt 0x0" ::: "memory");
#endif
}

__device__ __forceinline__ v8i lds_a_frag(const int8_t* base, int row0, int lane) {
  const int row = row0 + (lane & 15);
  const int kb = (lane >> 4) << 4;
  const int8_t* p = base + row * TPAD + kb;
  uint64_t d0 = *(const uint64_t*)(p);
  uint64_t d1 = *(const uint64_t*)(p + 8);
  uint64_t d2 = *(const uint64_t*)(p + 32);
  uint64_t d3 = *(const uint64_t*)(p + 40);
  v8i r;
  r[0] = (int)(uint32_t)d0; r[1] = (int)(uint32_t)(d0 >> 32);
  r[2] = (int)(uint32_t)d1; r[3] = (int)(uint32_t)(d1 >> 32);
  r[4] = (int)(uint32_t)d2; r[5] = (int)(uint32_t)(d2 >> 32);
  r[6] = (int)(uint32_t)d3; r[7] = (int)(uint32_t)(d3 >> 32);
  return r;
}

__device__ __forceinline__ v8i lds_b_frag(const int8_t* base, int row0, int lane) {
  const int row = row0 + (lane & 15);
  const int kb = (lane >> 4) << 4;
  const int8_t* p = base + row * TPAD + kb;
  v4i lo = *(const v4i*)(p);
  v4i hi = *(const v4i*)(p + 32);
  v8i r;
  r[0] = lo[0]; r[1] = lo[1]; r[2] = lo[2]; r[3] = lo[3];
  r[4] = hi[0]; r[5] = hi[1]; r[6] = hi[2]; r[7] = hi[3];
  return r;
}

template <int N>
__global__ __launch_bounds__(256) void fwht_quant_kernel(
    const float* __restrict__ x, int8_t* __restrict__ q,
    float* __restrict__ inv_scale) {
  __shared__ float s[N];
  __shared__ float red[256];
  const int t = threadIdx.x;
  const float* row = x + (size_t)blockIdx.x * N;
  for (int i = t; i < N; i += 256) s[i] = row[i];
  __syncthreads();
  for (int len = 1; len < N; len <<= 1) {
    for (int i = t; i < N / 2; i += 256) {
      const int lo = i & (len - 1);
      const int a  = ((i - lo) << 1) + lo;
      const int b  = a + len;
      const float xa = s[a], xb = s[b];
      s[a] = xa + xb;
      s[b] = xa - xb;
    }
    __syncthreads();
  }
  const float norm = 1.0f / sqrtf((float)N);
  float amax = 0.0f;
  for (int i = t; i < N; i += 256) {
    const float v = s[i] * norm;
    s[i] = v;
    amax = fmaxf(amax, fabsf(v));
  }
  red[t] = amax;
  __syncthreads();
  for (int w = 128; w > 0; w >>= 1) {
    if (t < w) red[t] = fmaxf(red[t], red[t + w]);
    __syncthreads();
  }
  const float a = fmaxf(red[0], EPSQ);
  const float scale = 127.0f / a;
  int8_t* qrow = q + (size_t)blockIdx.x * N;
  for (int i = t; i < N; i += 256) {
    float v = rintf(s[i] * scale);
    v = fminf(fmaxf(v, -128.0f), 127.0f);
    qrow[i] = (int8_t)v;
  }
  if (t == 0) inv_scale[blockIdx.x] = a / 127.0f;
}

__global__ __launch_bounds__(256) void abs_sum_partial_kernel(
    const float* __restrict__ w, size_t n, float* __restrict__ partial) {
  __shared__ float red[256];
  const size_t stride = (size_t)gridDim.x * 256;
  float acc = 0.0f;
  for (size_t i = (size_t)blockIdx.x * 256 + threadIdx.x; i < n; i += stride)
    acc += fabsf(w[i]);
  red[threadIdx.x] = acc;
  __syncthreads();
  for (int s = 128; s > 0; s >>= 1) {
    if (threadIdx.x < s) red[threadIdx.x] += red[threadIdx.x + s];
    __syncthreads();
  }
  if (threadIdx.x == 0) partial[blockIdx.x] = red[0];
}

__global__ __launch_bounds__(256) void finalize_scale_kernel(
    const float* __restrict__ partial, int nparts, float inv_count,
    float* __restrict__ scale_out) {
  __shared__ float red[256];
  float acc = 0.0f;
  for (int i = threadIdx.x; i < nparts; i += 256) acc += partial[i];
  red[threadIdx.x] = acc;
  __syncthreads();
  for (int s = 128; s > 0; s >>= 1) {
    if (threadIdx.x < s) red[threadIdx.x] += red[threadIdx.x + s];
    __syncthreads();
  }
  if (threadIdx.x == 0) scale_out[0] = fmaxf(red[0] * inv_count, EPSQ);
}

__global__ __launch_bounds__(256) void quant_weight_kernel(
    const float* __restrict__ w, size_t n, const float* __restrict__ scale,
    int8_t* __restrict__ q) {
  const float inv = 1.0f / scale[0];
  const size_t stride = (size_t)gridDim.x * 256;
  for (size_t i = (size_t)blockIdx.x * 256 + threadIdx.x; i < n; i += stride) {
    float v = rintf(w[i] * inv);
    v = fminf(fmaxf(v, -1.0f), 1.0f);
    q[i] = (int8_t)v;
  }
}

__global__ __launch_bounds__(256) void gemm_gateup_silu_kernel(
    const int8_t* __restrict__ A, const int8_t* __restrict__ Wg,
    const int8_t* __restrict__ Wu, const float* __restrict__ inv_as,
    const float* __restrict__ sc_g, const float* __restrict__ sc_u,
    float* __restrict__ Hout, int M, int N, int K) {
  __shared__ int8_t lsA[2 * 128 * TPAD];
  __shared__ int8_t lsBg[2 * 128 * TPAD];
  __shared__ int8_t lsBu[2 * 128 * TPAD];
  const int t = threadIdx.x;
  const int lane = t & 31;
  const int wave = t >> 5;
  const int wm = wave >> 2;
  const int wn = wave & 3;
  const int mblk = blockIdx.y * 128;
  const int nblk = blockIdx.x * 128;
  const int m0w = wm * 64;
  const int n0w = wn * 32;

  v8i accg[4][2] = {};
  v8i accu[4][2] = {};

  auto issue = [&](int buf, int k0) {
#pragma unroll
    for (int rep = 0; rep < 2; ++rep) {
      const int idx = t + rep * 256;
      const int row = idx >> 2, seg = (idx & 3) * 16;
      async_copy_b128(A + (size_t)(mblk + row) * K + k0 + seg,
                      &lsA[buf * (128 * TPAD) + row * TPAD + seg]);
    }
#pragma unroll
    for (int rep = 0; rep < 2; ++rep) {
      const int idx = t + rep * 256;
      const int row = idx >> 2, seg = (idx & 3) * 16;
      async_copy_b128(Wg + (size_t)(nblk + row) * K + k0 + seg,
                      &lsBg[buf * (128 * TPAD) + row * TPAD + seg]);
    }
#pragma unroll
    for (int rep = 0; rep < 2; ++rep) {
      const int idx = t + rep * 256;
      const int row = idx >> 2, seg = (idx & 3) * 16;
      async_copy_b128(Wu + (size_t)(nblk + row) * K + k0 + seg,
                      &lsBu[buf * (128 * TPAD) + row * TPAD + seg]);
    }
  };

  issue(0, 0);
  for (int k0 = 0; k0 < K; k0 += 64) {
    const int cur = (k0 >> 6) & 1;
    if (k0 + 64 < K) {
      issue(cur ^ 1, k0 + 64);
      wait_async_le6();
    } else {
      wait_async_le0();
    }
    __syncthreads();
    const int8_t* baA = &lsA[cur * (128 * TPAD)];
    const int8_t* baG = &lsBg[cur * (128 * TPAD)];
    const int8_t* baU = &lsBu[cur * (128 * TPAD)];
    v8i a[4], bg[2], bu[2];
#pragma unroll
    for (int i = 0; i < 4; ++i) a[i] = lds_a_frag(baA, m0w + 16 * i, lane);
#pragma unroll
    for (int j = 0; j < 2; ++j) {
      bg[j] = lds_b_frag(baG, n0w + 16 * j, lane);
      bu[j] = lds_b_frag(baU, n0w + 16 * j, lane);
    }
#pragma unroll
    for (int i = 0; i < 4; ++i) {
#pragma unroll
      for (int j = 0; j < 2; ++j) {
        accg[i][j] = __builtin_amdgcn_wmma_i32_16x16x64_iu8(
            true, a[i], true, bg[j], accg[i][j], false, false);
        accu[i][j] = __builtin_amdgcn_wmma_i32_16x16x64_iu8(
            true, a[i], true, bu[j], accu[i][j], false, false);
      }
    }
    __syncthreads();
  }

  const float wsg = sc_g[0];
  const float wsu = sc_u[0];
  const int lrow = (lane >> 4) << 3;
  const int lcol = lane & 15;
#pragma unroll
  for (int i = 0; i < 4; ++i) {
#pragma unroll
    for (int r = 0; r < 8; ++r) {
      const int m = mblk + m0w + 16 * i + lrow + r;
      const float as = inv_as[m];
#pragma unroll
      for (int j = 0; j < 2; ++j) {
        const int n = nblk + n0w + 16 * j + lcol;
        const float g = (float)accg[i][j][r] * as * wsg;
        const float u = (float)accu[i][j][r] * as * wsu;
        const float sg = g / (1.0f + __expf(-g));
        Hout[(size_t)m * N + n] = sg * u;
      }
    }
  }
}

__global__ __launch_bounds__(256) void gemm_iu8_kernel(
    const int8_t* __restrict__ A, const int8_t* __restrict__ W,
    const float* __restrict__ inv_as, const float* __restrict__ sc_w,
    float* __restrict__ out, int M, int N, int K) {
  __shared__ int8_t lsA[2 * 128 * TPAD];
  __shared__ int8_t lsB[2 * 256 * TPAD];
  const int t = threadIdx.x;
  const int lane = t & 31;
  const int wave = t >> 5;
  const int wm = wave >> 2;
  const int wn = wave & 3;
  const int mblk = blockIdx.y * 128;
  const int nblk = blockIdx.x * 256;
  const int m0w = wm * 64;
  const int n0w = wn * 64;

  v8i acc[4][4] = {};

  auto issue = [&](int buf, int k0) {
#pragma unroll
    for (int rep = 0; rep < 2; ++rep) {
      const int idx = t + rep * 256;
      const int row = idx >> 2, seg = (idx & 3) * 16;
      async_copy_b128(A + (size_t)(mblk + row) * K + k0 + seg,
                      &lsA[buf * (128 * TPAD) + row * TPAD + seg]);
    }
#pragma unroll
    for (int rep = 0; rep < 4; ++rep) {
      const int idx = t + rep * 256;
      const int row = idx >> 2, seg = (idx & 3) * 16;
      async_copy_b128(W + (size_t)(nblk + row) * K + k0 + seg,
                      &lsB[buf * (256 * TPAD) + row * TPAD + seg]);
    }
  };

  issue(0, 0);
  for (int k0 = 0; k0 < K; k0 += 64) {
    const int cur = (k0 >> 6) & 1;
    if (k0 + 64 < K) {
      issue(cur ^ 1, k0 + 64);
      wait_async_le6();
    } else {
      wait_async_le0();
    }
    __syncthreads();
    const int8_t* baA = &lsA[cur * (128 * TPAD)];
    const int8_t* baB = &lsB[cur * (256 * TPAD)];
    v8i a[4], b[4];
#pragma unroll
    for (int i = 0; i < 4; ++i) a[i] = lds_a_frag(baA, m0w + 16 * i, lane);
#pragma unroll
    for (int j = 0; j < 4; ++j) b[j] = lds_b_frag(baB, n0w + 16 * j, lane);
#pragma unroll
    for (int i = 0; i < 4; ++i) {
#pragma unroll
      for (int j = 0; j < 4; ++j) {
        acc[i][j] = __builtin_amdgcn_wmma_i32_16x16x64_iu8(
            true, a[i], true, b[j], acc[i][j], false, false);
      }
    }
    __syncthreads();
  }

  const float ws = sc_w[0];
  const int lrow = (lane >> 4) << 3;
  const int lcol = lane & 15;
#pragma unroll
  for (int i = 0; i < 4; ++i) {
#pragma unroll
    for (int r = 0; r < 8; ++r) {
      const int m = mblk + m0w + 16 * i + lrow + r;
      const float dq = inv_as[m] * ws;
#pragma unroll
      for (int j = 0; j < 4; ++j) {
        const int n = nblk + n0w + 16 * j + lcol;
        out[(size_t)m * N + n] = (float)acc[i][j][r] * dq;
      }
    }
  }
}

extern "C" void kernel_launch(void* const* d_in, const int* in_sizes, int n_in,
                              void* d_out, int out_size, void* d_ws,
                              size_t ws_size, hipStream_t stream) {
  const float* x      = (const float*)d_in[0];
  const float* w_gate = (const float*)d_in[1];
  const float* w_up   = (const float*)d_in[2];
  const float* w_down = (const float*)d_in[3];
  float* out = (float*)d_out;

  const int H = HDIM;
  const int I = IDIM;
  const int M = in_sizes[0] / H;
  const size_t IH = (size_t)I * H;

  uint8_t* ws = (uint8_t*)d_ws;
  size_t off = 0;
  auto alloc = [&](size_t bytes) -> uint8_t* {
    uint8_t* p = ws + off;
    off = (off + bytes + 255) & ~(size_t)255;
    return p;
  };
  int8_t* wq_g = (int8_t*)alloc(IH);
  int8_t* wq_u = (int8_t*)alloc(IH);
  int8_t* wq_d = (int8_t*)alloc(IH);
  int8_t* xq1  = (int8_t*)alloc((size_t)M * H);
  int8_t* xq2  = (int8_t*)alloc((size_t)M * I);
  float*  ias1 = (float*)alloc((size_t)M * sizeof(float));
  float*  ias2 = (float*)alloc((size_t)M * sizeof(float));
  float*  part = (float*)alloc(1024 * sizeof(float));
  float*  scg  = (float*)alloc(sizeof(float));
  float*  scu  = (float*)alloc(sizeof(float));
  float*  scd  = (float*)alloc(sizeof(float));
  float*  hbuf = (float*)alloc((size_t)M * I * sizeof(float));
  (void)ws_size; (void)n_in; (void)out_size;

  fwht_quant_kernel<HDIM><<<M, 256, 0, stream>>>(x, xq1, ias1);

  abs_sum_partial_kernel<<<1024, 256, 0, stream>>>(w_gate, IH, part);
  finalize_scale_kernel<<<1, 256, 0, stream>>>(part, 1024, 1.0f / (float)IH, scg);
  quant_weight_kernel<<<4096, 256, 0, stream>>>(w_gate, IH, scg, wq_g);

  abs_sum_partial_kernel<<<1024, 256, 0, stream>>>(w_up, IH, part);
  finalize_scale_kernel<<<1, 256, 0, stream>>>(part, 1024, 1.0f / (float)IH, scu);
  quant_weight_kernel<<<4096, 256, 0, stream>>>(w_up, IH, scu, wq_u);

  abs_sum_partial_kernel<<<1024, 256, 0, stream>>>(w_down, IH, part);
  finalize_scale_kernel<<<1, 256, 0, stream>>>(part, 1024, 1.0f / (float)IH, scd);
  quant_weight_kernel<<<4096, 256, 0, stream>>>(w_down, IH, scd, wq_d);

  gemm_gateup_silu_kernel<<<dim3(I / 128, M / 128), 256, 0, stream>>>(
      xq1, wq_g, wq_u, ias1, scg, scu, hbuf, M, I, H);

  fwht_quant_kernel<IDIM><<<M, 256, 0, stream>>>(hbuf, xq2, ias2);

  gemm_iu8_kernel<<<dim3(H / 256, M / 128), 256, 0, stream>>>(
      xq2, wq_d, ias2, scd, out, M, H, I);
}
